// GlobalLocalBlock_824633721375
// MI455X (gfx1250) — matrix operand order repaired and checked by reading, not compiled, not run on hardware
//
#include <hip/hip_runtime.h>
#include <hip/hip_bf16.h>
#include <math.h>

// ---------------------------------------------------------------------------
// GlobalLocalBlock for MI455X (gfx1250).
// All GEMM-shaped work on v_wmma_f32_16x16x32_f16 (f32 accumulate) with
// LDS-staged 128x64 block tiles; elementwise/softmax/LN on VALU+LDS.
// ---------------------------------------------------------------------------

typedef __attribute__((ext_vector_type(16))) _Float16 v16h;
typedef __attribute__((ext_vector_type(8)))  _Float16 v8h;
typedef __attribute__((ext_vector_type(8)))  float    v8f;

// ------------------------- f32 -> f16 convert ------------------------------
__global__ void glb_cvt_f16(const float* __restrict__ in, _Float16* __restrict__ out, int n) {
    int i = blockIdx.x * blockDim.x + threadIdx.x;
    if (i < n) out[i] = (_Float16)in[i];
}

// ------------------------- generic WMMA GEMM -------------------------------
// out[z][m][n] = alpha * sum_k A(m,k)*B(k,n) + bias[m]   (ACT==1: Mish)
// TA==0: A(m,k)=A[m*lda+k]          TA==1: A(m,k)=A[k*lda+m]
// TB==0: B(k,n)=B[k*ldb+n]          TB==1: B(k,n)=B[n*ldb+k]
// Block tile: 128(M) x 64(N) x 32(K); 8 waves, each wave 32x32 (4 wmma/step).
#define BK   32
#define LDS_STRIDE 40   // padded f16 stride: 80B rows -> 16B aligned, conflict-reduced

template<int TA, int TB, int ACT>
__global__ __launch_bounds__(256)
void glb_gemm_wmma(const _Float16* __restrict__ A, const _Float16* __restrict__ Bm,
                   const float* __restrict__ bias,
                   float* __restrict__ outF, _Float16* __restrict__ outH,
                   int M, int N, int K, int lda, int ldb, int ldo,
                   long long strideA, long long strideB, long long strideO,
                   float alpha)
{
    __shared__ _Float16 sA[128 * LDS_STRIDE];   // [m][k]
    __shared__ _Float16 sB[64  * LDS_STRIDE];   // [n][k]  (transposed B)

    const int z = blockIdx.z;
    A  += (size_t)z * (size_t)strideA;
    Bm += (size_t)z * (size_t)strideB;

    const int t    = threadIdx.x;
    const int lane = t & 31;
    const int wave = t >> 5;
    const int m0blk = blockIdx.x * 128;
    const int n0blk = blockIdx.y * 64;
    const int wm = (wave >> 1) * 32;            // 4 waves along M
    const int wn = (wave & 1) * 32;             // 2 waves along N

    // ISA 16-bit fragment layouts (wave32)
    const int arow = lane & 15;
    const int half = lane >> 4;
    const int akb  = half * 8;                  // A: K = akb+j , 16+akb+j
    const int bkb  = half * 8;                  // B: K = bkb+j , 16+bkb+j
    const int col16 = lane & 15;

    v8f acc[2][2] = {{{}, {}}, {{}, {}}};

    for (int k0 = 0; k0 < K; k0 += BK) {
        __syncthreads();   // protect previous tile until all waves consumed it

        // ---- prefetch next k-tile while this one is being staged/consumed ----
        if (k0 + BK < K) {
            if (TA == 0)  __builtin_prefetch(A + (size_t)(m0blk + (t >> 2)) * lda + (k0 + BK), 0, 1);
            else          __builtin_prefetch(A + (size_t)(k0 + BK + (t >> 4)) * lda + m0blk, 0, 1);
            if (TB == 0)  __builtin_prefetch(Bm + (size_t)(k0 + BK + (t >> 3)) * ldb + n0blk, 0, 1);
            else          __builtin_prefetch(Bm + (size_t)(n0blk + (t >> 2)) * ldb + (k0 + BK), 0, 1);
        }

        // ---- stage A tile: 128x32 f16 (2 passes of 256 x b128) ----
        if (TA == 0) {
            #pragma unroll
            for (int p = 0; p < 2; ++p) {
                const int c   = p * 256 + t;
                const int row = c >> 2;             // 4 b128 chunks per row
                const int kc  = (c & 3) * 8;
                *(v8h*)&sA[row * LDS_STRIDE + kc] =
                    *(const v8h*)(A + (size_t)(m0blk + row) * lda + k0 + kc);
            }
        } else {
            #pragma unroll
            for (int p = 0; p < 2; ++p) {
                const int c  = p * 256 + t;
                const int kk = c >> 4;              // 16 m-chunks per k line
                const int mc = (c & 15) * 8;
                const v8h v = *(const v8h*)(A + (size_t)(k0 + kk) * lda + m0blk + mc);
                #pragma unroll
                for (int j = 0; j < 8; ++j) sA[(mc + j) * LDS_STRIDE + kk] = v[j];
            }
        }

        // ---- stage B tile: 32x64 f16, stored transposed sB[n][k] ----
        if (TB == 0) {
            const int kk = t >> 3;                  // 8 n-chunks per k line
            const int nc = (t & 7) * 8;
            const v8h v = *(const v8h*)(Bm + (size_t)(k0 + kk) * ldb + n0blk + nc);
            #pragma unroll
            for (int j = 0; j < 8; ++j) sB[(nc + j) * LDS_STRIDE + kk] = v[j];
        } else {
            const int nn = t >> 2;                  // 4 k-chunks per n line
            const int kc = (t & 3) * 8;
            *(v8h*)&sB[nn * LDS_STRIDE + kc] =
                *(const v8h*)(Bm + (size_t)(n0blk + nn) * ldb + k0 + kc);
        }

        __syncthreads();

        // ---- fragments from LDS (16B-aligned ds_load_b128) + 4 WMMAs ----
        v16h afr[2], bfr[2];
        #pragma unroll
        for (int mi = 0; mi < 2; ++mi) {
            const _Float16* pa = &sA[(wm + mi * 16 + arow) * LDS_STRIDE];
            const v8h lo = *(const v8h*)&pa[akb];
            const v8h hi = *(const v8h*)&pa[16 + akb];
            #pragma unroll
            for (int j = 0; j < 8; ++j) { afr[mi][j] = lo[j]; afr[mi][8 + j] = hi[j]; }
        }
        #pragma unroll
        for (int ni = 0; ni < 2; ++ni) {
            const _Float16* pb = &sB[(wn + ni * 16 + col16) * LDS_STRIDE + bkb];
            const v8h lo = *(const v8h*)pb;
            const v8h hi = *(const v8h*)(pb + 16);
            #pragma unroll
            for (int j = 0; j < 8; ++j) { bfr[ni][j] = lo[j]; bfr[ni][8 + j] = hi[j]; }
        }
        #pragma unroll
        for (int mi = 0; mi < 2; ++mi)
            #pragma unroll
            for (int ni = 0; ni < 2; ++ni)
                acc[mi][ni] = __builtin_amdgcn_wmma_f32_16x16x32_f16(
                    false, afr[mi], false, bfr[ni], (short)0, acc[mi][ni], false, false);
    }

    // ---- epilogue: C/D layout -> M = i + 8*half, N = lane&15 ----
    const long long zoff = (long long)z * strideO;
    #pragma unroll
    for (int mi = 0; mi < 2; ++mi) {
        #pragma unroll
        for (int ni = 0; ni < 2; ++ni) {
            #pragma unroll
            for (int i = 0; i < 8; ++i) {
                const int m = m0blk + wm + mi * 16 + i + 8 * half;
                const int n = n0blk + wn + ni * 16 + col16;
                float v = acc[mi][ni][i] * alpha + (bias ? bias[m] : 0.0f);
                if (ACT == 1) {  // Mish
                    float sp = (v > 20.0f) ? v : log1pf(expf(v));
                    v = v * tanhf(sp);
                }
                const size_t off = (size_t)zoff + (size_t)m * (size_t)ldo + n;
                if (outF) outF[off] = v;
                if (outH) outH[off] = (_Float16)v;
            }
        }
    }
}

// ------------------- per-position channel mean of g ------------------------
__global__ void glb_gmean(const _Float16* __restrict__ g16, float* __restrict__ gmean,
                          int B, int C, int L)
{
    const int lane = threadIdx.x & 31;
    const int wave = threadIdx.x >> 5;
    const int j = blockIdx.x * (blockDim.x >> 5) + wave;
    if (j >= B * L) return;
    const int b = j / L, pos = j % L;
    const _Float16* gb = g16 + (size_t)b * C * L + pos;
    float s = 0.0f;
    for (int c = lane; c < C; c += 32) s += (float)gb[(size_t)c * L];
    #pragma unroll
    for (int o = 16; o > 0; o >>= 1) s += __shfl_xor(s, o, 32);
    if (lane == 0) gmean[j] = s / (float)C;
}

// ------------------- local-attention gate ----------------------------------
__global__ void glb_gate(const float* __restrict__ pg, const float* __restrict__ gmean,
                         const float* __restrict__ wlw, const float* __restrict__ wlb,
                         float* __restrict__ gate, int L, int W)
{
    const int bl = blockIdx.x;           // b*L + l
    const int b = bl / L, l = bl % L;
    const int w = threadIdx.x;           // 0..63
    const int idx = l + w - W / 2;
    const bool valid = (idx >= 0 && idx < L);
    const float s = valid ? pg[((size_t)b * L + l) * L + idx] : 0.0f;

    __shared__ float sh[64];
    sh[w] = s; __syncthreads();
    for (int o = 32; o > 0; o >>= 1) { if (w < o) sh[w] = fmaxf(sh[w], sh[w + o]); __syncthreads(); }
    const float mx = sh[0]; __syncthreads();

    const float e = expf(s - mx);
    sh[w] = e; __syncthreads();
    for (int o = 32; o > 0; o >>= 1) { if (w < o) sh[w] += sh[w + o]; __syncthreads(); }
    const float denom = sh[0]; __syncthreads();

    const float pr = e / denom;
    sh[w] = valid ? pr * gmean[b * L + idx] : 0.0f;
    __syncthreads();
    for (int o = 32; o > 0; o >>= 1) { if (w < o) sh[w] += sh[w + o]; __syncthreads(); }
    if (w == 0) {
        gate[bl] = 1.0f / (1.0f + expf(-(wlw[0] * sh[0] + wlb[0])));
    }
}

// ------------------- row softmax (global attention) ------------------------
__global__ void glb_softmax_rows(const float* __restrict__ gs, _Float16* __restrict__ gp16, int L)
{
    const int row = blockIdx.x;
    const int m = threadIdx.x;
    const float v = gs[(size_t)row * L + m];
    __shared__ float sh[256];
    sh[m] = v; __syncthreads();
    for (int o = 128; o > 0; o >>= 1) { if (m < o) sh[m] = fmaxf(sh[m], sh[m + o]); __syncthreads(); }
    const float mx = sh[0]; __syncthreads();
    const float e = expf(v - mx);
    sh[m] = e; __syncthreads();
    for (int o = 128; o > 0; o >>= 1) { if (m < o) sh[m] += sh[m + o]; __syncthreads(); }
    gp16[(size_t)row * L + m] = (_Float16)(e / sh[0]);
}

// ------------------- residual + gate + LayerNorm1 --------------------------
__global__ void glb_h_ln1(const float* __restrict__ wgout, const float* __restrict__ x,
                          const float* __restrict__ gate,
                          const float* __restrict__ ln1g, const float* __restrict__ ln1b,
                          float* __restrict__ h1, _Float16* __restrict__ h116,
                          int B, int C, int L)
{
    const int r = blockIdx.x;            // r = l*B + b
    const int l = r / B, b = r % B;
    const int R = B * L;
    const float gt = gate[b * L + l];
    const int c0 = threadIdx.x, c1 = threadIdx.x + 256;

    const float y0 = wgout[((size_t)b * C + c0) * L + l] * gt + x[((size_t)b * C + c0) * L + l];
    const float y1 = wgout[((size_t)b * C + c1) * L + l] * gt + x[((size_t)b * C + c1) * L + l];

    __shared__ float sh[256];
    sh[threadIdx.x] = y0 + y1; __syncthreads();
    for (int o = 128; o > 0; o >>= 1) { if (threadIdx.x < o) sh[threadIdx.x] += sh[threadIdx.x + o]; __syncthreads(); }
    const float mean = sh[0] / (float)C; __syncthreads();
    const float d0 = y0 - mean, d1 = y1 - mean;
    sh[threadIdx.x] = d0 * d0 + d1 * d1; __syncthreads();
    for (int o = 128; o > 0; o >>= 1) { if (threadIdx.x < o) sh[threadIdx.x] += sh[threadIdx.x + o]; __syncthreads(); }
    const float inv = rsqrtf(sh[0] / (float)C + 1e-5f);

    const float v0 = d0 * inv * ln1g[c0] + ln1b[c0];
    const float v1 = d1 * inv * ln1g[c1] + ln1b[c1];
    h1[(size_t)c0 * R + r] = v0;  h116[(size_t)c0 * R + r] = (_Float16)v0;
    h1[(size_t)c1 * R + r] = v1;  h116[(size_t)c1 * R + r] = (_Float16)v1;
}

// ------------------- residual + LayerNorm2 + final transpose ---------------
__global__ void glb_final_ln2(const float* __restrict__ m2, const float* __restrict__ h1,
                              const float* __restrict__ ln2g, const float* __restrict__ ln2b,
                              float* __restrict__ out, int B, int C, int L)
{
    const int r = blockIdx.x;            // r = l*B + b
    const int l = r / B, b = r % B;
    const int R = B * L;
    const int c0 = threadIdx.x, c1 = threadIdx.x + 256;

    const float y0 = m2[(size_t)c0 * R + r] + h1[(size_t)c0 * R + r];
    const float y1 = m2[(size_t)c1 * R + r] + h1[(size_t)c1 * R + r];

    __shared__ float sh[256];
    sh[threadIdx.x] = y0 + y1; __syncthreads();
    for (int o = 128; o > 0; o >>= 1) { if (threadIdx.x < o) sh[threadIdx.x] += sh[threadIdx.x + o]; __syncthreads(); }
    const float mean = sh[0] / (float)C; __syncthreads();
    const float d0 = y0 - mean, d1 = y1 - mean;
    sh[threadIdx.x] = d0 * d0 + d1 * d1; __syncthreads();
    for (int o = 128; o > 0; o >>= 1) { if (threadIdx.x < o) sh[threadIdx.x] += sh[threadIdx.x + o]; __syncthreads(); }
    const float inv = rsqrtf(sh[0] / (float)C + 1e-5f);

    out[((size_t)b * C + c0) * L + l] = d0 * inv * ln2g[c0] + ln2b[c0];
    out[((size_t)b * C + c1) * L + l] = d1 * inv * ln2g[c1] + ln2b[c1];
}

// ===========================================================================
extern "C" void kernel_launch(void* const* d_in, const int* in_sizes, int n_in,
                              void* d_out, int out_size, void* d_ws, size_t ws_size,
                              hipStream_t stream) {
    const int B = 8, C = 512, L = 256, C4 = 2048;
    const int R = L * B;                       // 2048
    const float scale = 0.04419417382415922f;  // 512^-0.5

    const float* x       = (const float*)d_in[0];
    const float* theta_w = (const float*)d_in[1];
    const float* theta_b = (const float*)d_in[2];
    const float* phi_w   = (const float*)d_in[3];
    const float* phi_b   = (const float*)d_in[4];
    const float* g_w     = (const float*)d_in[5];
    const float* g_b     = (const float*)d_in[6];
    const float* wl_w    = (const float*)d_in[7];
    const float* wl_b    = (const float*)d_in[8];
    const float* wg_w    = (const float*)d_in[9];
    const float* wg_b    = (const float*)d_in[10];
    const float* conv1_w = (const float*)d_in[11];
    const float* conv1_b = (const float*)d_in[12];
    const float* conv2_w = (const float*)d_in[13];
    const float* conv2_b = (const float*)d_in[14];
    const float* ln1_g   = (const float*)d_in[15];
    const float* ln1_b   = (const float*)d_in[16];
    const float* ln2_g   = (const float*)d_in[17];
    const float* ln2_b   = (const float*)d_in[18];
    float* out = (float*)d_out;

    // ---- workspace carve-up (bump allocator, 256B aligned) ----
    char* wp = (char*)d_ws;
    auto alloc = [&](size_t bytes) -> char* {
        char* r = wp; wp += (bytes + 255) & ~(size_t)255; return r;
    };
    const size_t BCL = (size_t)B * C * L;      // 1,048,576
    const size_t BLL = (size_t)B * L * L;      //   524,288
    _Float16* x16    = (_Float16*)alloc(BCL * 2);
    _Float16* thw16  = (_Float16*)alloc((size_t)C * C * 2);
    _Float16* phw16  = (_Float16*)alloc((size_t)C * C * 2);
    _Float16* gw16   = (_Float16*)alloc((size_t)C * C * 2);
    _Float16* wgw16  = (_Float16*)alloc((size_t)C * C * 2);
    _Float16* c1w16  = (_Float16*)alloc((size_t)C4 * C * 2);
    _Float16* c2w16  = (_Float16*)alloc((size_t)C * C4 * 2);
    _Float16* th16   = (_Float16*)alloc(BCL * 2);
    _Float16* ph16   = (_Float16*)alloc(BCL * 2);
    _Float16* g16    = (_Float16*)alloc(BCL * 2);
    float*    gs     = (float*)   alloc(BLL * 4);
    float*    pg     = (float*)   alloc(BLL * 4);
    _Float16* gp16   = (_Float16*)alloc(BLL * 2);
    _Float16* gout16 = (_Float16*)alloc(BCL * 2);
    float*    wgout  = (float*)   alloc(BCL * 4);
    float*    gmean  = (float*)   alloc((size_t)B * L * 4);
    float*    gate   = (float*)   alloc((size_t)B * L * 4);
    float*    h1     = (float*)   alloc((size_t)C * R * 4);
    _Float16* h116   = (_Float16*)alloc((size_t)C * R * 2);
    _Float16* m116   = (_Float16*)alloc((size_t)C4 * R * 2);
    float*    m2     = (float*)   alloc((size_t)C * R * 4);

    // ---- 1) f32 -> f16 conversions ----
    auto cvt = [&](const float* src, _Float16* dst, int n) {
        glb_cvt_f16<<<(n + 255) / 256, 256, 0, stream>>>(src, dst, n);
    };
    cvt(x,       x16,   (int)BCL);
    cvt(theta_w, thw16, C * C);
    cvt(phi_w,   phw16, C * C);
    cvt(g_w,     gw16,  C * C);
    cvt(wg_w,    wgw16, C * C);
    cvt(conv1_w, c1w16, C4 * C);
    cvt(conv2_w, c2w16, C * C4);

    // ---- 2) theta/phi/g = W @ x + b   (M=C, N=L, K=C, batch=B) ----
    {
        dim3 grid(C / 128, L / 64, B);
        glb_gemm_wmma<0, 0, 0><<<grid, 256, 0, stream>>>(
            thw16, x16, theta_b, nullptr, th16,
            C, L, C, C, L, L, 0, (long long)C * L, (long long)C * L, 1.0f);
        glb_gemm_wmma<0, 0, 0><<<grid, 256, 0, stream>>>(
            phw16, x16, phi_b, nullptr, ph16,
            C, L, C, C, L, L, 0, (long long)C * L, (long long)C * L, 1.0f);
        glb_gemm_wmma<0, 0, 0><<<grid, 256, 0, stream>>>(
            gw16, x16, g_b, nullptr, g16,
            C, L, C, C, L, L, 0, (long long)C * L, (long long)C * L, 1.0f);
    }

    // ---- 3) gs = scale*phi^T theta ; pg = scale*phi^T g  (M=N=L, K=C) ----
    {
        dim3 grid(L / 128, L / 64, B);
        glb_gemm_wmma<1, 0, 0><<<grid, 256, 0, stream>>>(
            ph16, th16, nullptr, gs, nullptr,
            L, L, C, L, L, L, (long long)C * L, (long long)C * L, (long long)L * L, scale);
        glb_gemm_wmma<1, 0, 0><<<grid, 256, 0, stream>>>(
            ph16, g16, nullptr, pg, nullptr,
            L, L, C, L, L, L, (long long)C * L, (long long)C * L, (long long)L * L, scale);
    }

    // ---- 4) local gate path ----
    glb_gmean<<<(B * L + 7) / 8, 256, 0, stream>>>(g16, gmean, B, C, L);
    glb_gate<<<B * L, 64, 0, stream>>>(pg, gmean, wl_w, wl_b, gate, L, 64);

    // ---- 5) global softmax + apply: gout = g @ gp^T  (M=C, N=L, K=L) ----
    glb_softmax_rows<<<B * L, 256, 0, stream>>>(gs, gp16, L);
    {
        dim3 grid(C / 128, L / 64, B);
        glb_gemm_wmma<0, 1, 0><<<grid, 256, 0, stream>>>(
            g16, gp16, nullptr, nullptr, gout16,
            C, L, L, L, L, L, (long long)C * L, (long long)L * L, (long long)C * L, 1.0f);
        glb_gemm_wmma<0, 0, 0><<<grid, 256, 0, stream>>>(
            wgw16, gout16, wg_b, wgout, nullptr,
            C, L, C, C, L, L, 0, (long long)C * L, (long long)C * L, 1.0f);
    }

    // ---- 6) residual + gate + LN1 ----
    glb_h_ln1<<<R, 256, 0, stream>>>(wgout, x, gate, ln1_g, ln1_b, h1, h116, B, C, L);

    // ---- 7) MLP: m1 = Mish(conv1_w @ h1 + b1); m2 = conv2_w @ m1 + b2 ----
    {
        dim3 grid1(C4 / 128, R / 64, 1);
        glb_gemm_wmma<0, 0, 1><<<grid1, 256, 0, stream>>>(
            c1w16, h116, conv1_b, nullptr, m116,
            C4, R, C, C, R, R, 0, 0, 0, 1.0f);
        dim3 grid2(C / 128, R / 64, 1);
        glb_gemm_wmma<0, 0, 0><<<grid2, 256, 0, stream>>>(
            c2w16, m116, conv2_b, m2, nullptr,
            C, R, C4, C4, R, R, 0, 0, 0, 1.0f);
    }

    // ---- 8) residual + LN2 + transpose to [B,C,L] ----
    glb_final_ln2<<<R, 256, 0, stream>>>(m2, h1, ln2_g, ln2_b, out, B, C, L);
}
